// Dense_1322849927863
// MI455X (gfx1250) — matrix operand order repaired and checked by reading, not compiled, not run on hardware
//
#include <hip/hip_runtime.h>

typedef __attribute__((ext_vector_type(16))) _Float16 v16h;
typedef __attribute__((ext_vector_type(8)))  _Float16 v8h;
typedef __attribute__((ext_vector_type(4)))  _Float16 v4h;
typedef __attribute__((ext_vector_type(8)))  float    v8f;

#define B_  16
#define L_  12
#define T_  1024
#define F_  768
#define H_  256
#define NL_ 8

// LDS tile row pitch in halves: 776*2B = 1552B = 388 dwords ≡ 4 (mod 64 banks)
#define APITCH 776

static __device__ __forceinline__ v16h catv(v8h lo, v8h hi) {
  v16h r;
#pragma unroll
  for (int i = 0; i < 8; ++i) { r[i] = lo[i]; r[i + 8] = hi[i]; }
  return r;
}

// ---------------------------------------------------------------------------
// prep: f32->f16 weight conversion + zero the pooled accumulator (every call:
// ws is poisoned once and graph replays must be self-initializing).
// ---------------------------------------------------------------------------
__global__ void prep_kernel(const float* __restrict__ w1,
                            const float* __restrict__ w2,
                            _Float16* __restrict__ w1h,
                            _Float16* __restrict__ w2h,
                            float* __restrict__ pooled) {
  const int N1 = H_ * F_;   // 196608
  const int N2 = H_ * H_;   // 65536
  const int NP = B_ * H_;   // 4096
  int idx = blockIdx.x * 256 + threadIdx.x;
  if (idx < N1) {
    w1h[idx] = (_Float16)w1[idx];
  } else if (idx < N1 + N2) {
    w2h[idx - N1] = (_Float16)w2[idx - N1];
  } else if (idx < N1 + N2 + NP) {
    pooled[idx - N1 - N2] = 0.0f;
  }
}

// ---------------------------------------------------------------------------
// Fused: layer aggregation (HBM-bound, reads x once) + GEMM1 + bias + ReLU.
// One block = one 16-row M-tile (16 timesteps of one batch). 8 waves x 2
// N-tiles = all 256 output channels. 24 v_wmma_f32_16x16x32_f16 per tile.
// ---------------------------------------------------------------------------
__global__ void __launch_bounds__(256) gemm1_kernel(
    const float* __restrict__ x, const float* __restrict__ aggr_w,
    const _Float16* __restrict__ w1h, const float* __restrict__ b1,
    _Float16* __restrict__ h1) {
  __shared__ _Float16 sh[16 * APITCH];

  const int tile = blockIdx.x;        // 0..1023
  const int b    = tile >> 6;         // 64 tiles per batch (1024/16)
  const int t0   = (tile & 63) << 4;

  float aw[L_];
#pragma unroll
  for (int l = 0; l < L_; ++l) aw[l] = aggr_w[l];   // uniform -> s_loads

  // Phase 1: h[m][f] = sum_l aw[l]*x[b,l,t0+m,f], stored as f16 in LDS.
  const float4* x4 = (const float4*)x;
  const int rowq = F_ / 4;  // 192 float4 per row
  for (int i = threadIdx.x; i < 16 * rowq; i += 256) {
    int m  = i / rowq;
    int f4 = i % rowq;
    const float4* p = x4 + ((size_t)(b * L_) * T_ + (t0 + m)) * rowq + f4;
    float4 acc = make_float4(0.f, 0.f, 0.f, 0.f);
#pragma unroll
    for (int l = 0; l < L_; ++l) {
      float4 v = p[(size_t)l * T_ * rowq];
      acc.x += aw[l] * v.x; acc.y += aw[l] * v.y;
      acc.z += aw[l] * v.z; acc.w += aw[l] * v.w;
    }
    v4h hv;
    hv[0] = (_Float16)acc.x; hv[1] = (_Float16)acc.y;
    hv[2] = (_Float16)acc.z; hv[3] = (_Float16)acc.w;
    *(v4h*)(sh + m * APITCH + f4 * 4) = hv;
  }
  __syncthreads();

  // Phase 2: WMMA over K=768.
  const int lane = threadIdx.x & 31;
  const int wave = threadIdx.x >> 5;
  const int mn   = lane & 15;   // A row / B column / C column for this lane
  const int half = lane >> 4;
  const int rowbase = tile * 16;

#pragma unroll
  for (int nt0 = 0; nt0 < 2; ++nt0) {
    const int nt   = wave + nt0 * 8;
    const int ncol = nt * 16 + mn;
    const _Float16* wrow = w1h + ncol * F_;
    const _Float16* arow = sh + mn * APITCH;
    v8f c = {};
    for (int k0 = 0; k0 < F_; k0 += 32) {
      v8h a0  = *(const v8h*)(arow + k0 + half * 16);       // K = k0+half*16..+7
      v8h a1  = *(const v8h*)(arow + k0 + half * 16 + 8);   // ..+15
      v8h bb0 = *(const v8h*)(wrow + k0 + half * 16);       // K = k0+half*16..+7
      v8h bb1 = *(const v8h*)(wrow + k0 + half * 16 + 8);   // ..+15
      c = __builtin_amdgcn_wmma_f32_16x16x32_f16(
          false, catv(a0, a1), false, catv(bb0, bb1), (short)0, c, false, false);
    }
    const float bias = b1[ncol];
#pragma unroll
    for (int r = 0; r < 8; ++r) {
      float v = c[r] + bias;
      v = v > 0.f ? v : 0.f;                 // ReLU
      int mrow = r + half * 8;               // C layout: M = r + 8*(lane/16)
      h1[(rowbase + mrow) * H_ + ncol] = (_Float16)v;
    }
  }
}

// ---------------------------------------------------------------------------
// Fused: GEMM2 + bias + ReLU + masked column-sum pooling (atomic accumulate).
// ---------------------------------------------------------------------------
__global__ void __launch_bounds__(256) gemm2_kernel(
    const _Float16* __restrict__ h1, const _Float16* __restrict__ w2h,
    const float* __restrict__ b2, const int* __restrict__ lengths,
    float* __restrict__ pooled) {
  const int tile = blockIdx.x;
  const int b    = tile >> 6;
  const int t0   = (tile & 63) << 4;
  const int len  = lengths[b];

  const int lane = threadIdx.x & 31;
  const int wave = threadIdx.x >> 5;
  const int mn   = lane & 15;
  const int half = lane >> 4;
  const int rowbase = tile * 16;

#pragma unroll
  for (int nt0 = 0; nt0 < 2; ++nt0) {
    const int nt   = wave + nt0 * 8;
    const int ncol = nt * 16 + mn;
    const _Float16* wrow = w2h + ncol * H_;
    const _Float16* arow = h1 + (size_t)(rowbase + mn) * H_;
    v8f c = {};
#pragma unroll
    for (int k0 = 0; k0 < H_; k0 += 32) {
      v8h a0  = *(const v8h*)(arow + k0 + half * 16);
      v8h a1  = *(const v8h*)(arow + k0 + half * 16 + 8);
      v8h bb0 = *(const v8h*)(wrow + k0 + half * 16);
      v8h bb1 = *(const v8h*)(wrow + k0 + half * 16 + 8);
      c = __builtin_amdgcn_wmma_f32_16x16x32_f16(
          false, catv(a0, a1), false, catv(bb0, bb1), (short)0, c, false, false);
    }
    const float bias = b2[ncol];
    float colsum = 0.f;
#pragma unroll
    for (int r = 0; r < 8; ++r) {
      float v = c[r] + bias;
      v = v > 0.f ? v : 0.f;
      int mrow = r + half * 8;
      if (t0 + mrow < len) colsum += v;     // masked pooling (t < lengths[b])
    }
    // combine the two lane-halves that hold the same column (M 0-7 vs 8-15)
    colsum += __shfl_xor(colsum, 16, 32);
    if (half == 0) atomicAdd(&pooled[b * H_ + ncol], colsum);
  }
}

// ---------------------------------------------------------------------------
// Head: out[b,j] = (pooled[b,:]/len_b) . w3[j,:] + b3[j]   (16x8 outputs)
// ---------------------------------------------------------------------------
__global__ void head_kernel(const float* __restrict__ pooled,
                            const int* __restrict__ lengths,
                            const float* __restrict__ w3,
                            const float* __restrict__ b3,
                            float* __restrict__ out) {
  int i = threadIdx.x;
  if (i >= B_ * NL_) return;
  int b = i >> 3, j = i & 7;
  float inv = 1.0f / (float)lengths[b];
  float s = 0.f;
  for (int g = 0; g < H_; ++g) s += pooled[b * H_ + g] * w3[j * H_ + g];
  out[i] = s * inv + b3[j];
}

// ---------------------------------------------------------------------------
extern "C" void kernel_launch(void* const* d_in, const int* in_sizes, int n_in,
                              void* d_out, int out_size, void* d_ws, size_t ws_size,
                              hipStream_t stream) {
  const float* x       = (const float*)d_in[0];
  const int*   lengths = (const int*)d_in[1];
  const float* aggr_w  = (const float*)d_in[2];
  const float* w1      = (const float*)d_in[3];
  const float* b1      = (const float*)d_in[4];
  const float* w2      = (const float*)d_in[5];
  const float* b2      = (const float*)d_in[6];
  const float* w3      = (const float*)d_in[7];
  const float* b3      = (const float*)d_in[8];
  float* out = (float*)d_out;

  char* ws = (char*)d_ws;
  _Float16* w1h    = (_Float16*)(ws + 0);        // 256*768*2  = 393216 B
  _Float16* w2h    = (_Float16*)(ws + 393216);   // 256*256*2  = 131072 B
  _Float16* h1     = (_Float16*)(ws + 524288);   // 16384*256*2 = 8388608 B
  float*    pooled = (float*)   (ws + 8912896);  // 16*256*4   = 16384 B

  prep_kernel <<<1040, 256, 0, stream>>>(w1, w2, w1h, w2h, pooled);
  gemm1_kernel<<<1024, 256, 0, stream>>>(x, aggr_w, w1h, b1, h1);
  gemm2_kernel<<<1024, 256, 0, stream>>>(h1, w2h, b2, lengths, pooled);
  head_kernel <<<1, 128, 0, stream>>>(pooled, lengths, w3, b3, out);
}
